// EncoderLayer_68994354642952
// MI455X (gfx1250) — matrix operand order repaired and checked by reading, not compiled, not run on hardware
//
#include <hip/hip_runtime.h>
#include <hip/hip_bf16.h>

// ---------------------------------------------------------------------------
// Types
// ---------------------------------------------------------------------------
typedef _Float16 h16;
typedef __attribute__((ext_vector_type(16))) _Float16 v16h;
typedef __attribute__((ext_vector_type(8)))  _Float16 v8h;
typedef __attribute__((ext_vector_type(8)))  float    v8f;

union U16 { v16h v; v8h h[2]; };

// ---------------------------------------------------------------------------
// Problem constants
// ---------------------------------------------------------------------------
#define EDIM 512
#define HHEADS 8
#define RDIM 128
#define FFDIM 2048
#define BB 8
#define SS 1024
#define TOK (BB*SS)          // 8192 tokens
#define HE (HHEADS*EDIM)     // 4096

// GEMM tiling
#define BM 64
#define BN 128
#define BK 64
#define LDK 80               // padded LDS stride (halves); 160B rows, 16B aligned

// ---------------------------------------------------------------------------
// CDNA5 async global->LDS copies (CDNA5 ISA).
// IOFFSET is added to BOTH the global and LDS addresses, so one VADDR/VDST
// pair covers a thread's whole contiguous slice. Tracked by ASYNCcnt.
// LDS VDST VGPR holds the wave-relative LDS byte address = low 32 bits of the
// generic pointer (LDS aperture maps addr[31:0]).
// ---------------------------------------------------------------------------
__device__ __forceinline__ unsigned lds_addr(const void* p) {
  return (unsigned)(unsigned long long)p;
}
__device__ __forceinline__ void async_copy_32B(unsigned lds, const h16* g) {
  asm volatile("global_load_async_to_lds_b128 %0, %1, off\n\t"
               "global_load_async_to_lds_b128 %0, %1, off offset:16"
               :: "v"(lds), "v"(g) : "memory");
}
__device__ __forceinline__ void async_copy_64B(unsigned lds, const h16* g) {
  asm volatile("global_load_async_to_lds_b128 %0, %1, off\n\t"
               "global_load_async_to_lds_b128 %0, %1, off offset:16\n\t"
               "global_load_async_to_lds_b128 %0, %1, off offset:32\n\t"
               "global_load_async_to_lds_b128 %0, %1, off offset:48"
               :: "v"(lds), "v"(g) : "memory");
}

// ---------------------------------------------------------------------------
// WMMA fragment loads from LDS (per CDNA5 ISA layouts)
// A 16x32 f16: lanes 0-15 = rows, K split {g*8+0..7, 16+g*8+0..7}
// B 32x16 f16 (stored transposed, N-major): lane%16 = col, K split {g*8+0..7, 16+g*8+0..7}
// ---------------------------------------------------------------------------
__device__ __forceinline__ v16h frag_a(const h16* t, int kb, int lane) {
  const int g = lane >> 4, m = lane & 15;
  const h16* p = t + m * LDK + kb + g * 8;
  U16 u; u.h[0] = *(const v8h*)p; u.h[1] = *(const v8h*)(p + 16);
  return u.v;
}
__device__ __forceinline__ v16h frag_b(const h16* t, int kb, int lane) {
  const int g = lane >> 4, n = lane & 15;
  const h16* p = t + n * LDK + kb + g * 8;
  U16 u; u.h[0] = *(const v8h*)p; u.h[1] = *(const v8h*)(p + 16);
  return u.v;
}

// ---------------------------------------------------------------------------
// Generic batched WMMA GEMM:  C[z] = A[z] (MxK) * Bt[z]^T (Bt is NxK) + bias
// Double-buffered LDS, async global->LDS copies, WMMA f16->f32.
// mode 0: f32 out, mode 1: f16 out, mode 2: f16 transposed out (C^T[N][ldc])
// rowScale (optional): multiply output row m by rowScale[m] (after bias)
// batch index z = z1*z2count + z2 with independent strides.
// ---------------------------------------------------------------------------
__global__ __launch_bounds__(256, 2)
void wmma_gemm_kernel(const h16* __restrict__ A, const h16* __restrict__ Bt,
                      const float* __restrict__ bias,
                      const float* __restrict__ rowScale,
                      void* __restrict__ C,
                      int M, int N, int K, int lda, int ldb, int ldc,
                      long sA1, long sA2, long sB1, long sB2,
                      long sC1, long sC2, long sBias, int z2count, int mode)
{
  (void)M; (void)N;
  const int z  = blockIdx.z;
  const int z1 = z / z2count, z2 = z - z1 * z2count;
  A  += (long)z1 * sA1 + (long)z2 * sA2;
  Bt += (long)z1 * sB1 + (long)z2 * sB2;
  const long cOff = (long)z1 * sC1 + (long)z2 * sC2;
  if (bias) bias += (long)z1 * sBias;

  __shared__ h16 As[2][BM][LDK];
  __shared__ h16 Bs[2][BN][LDK];

  const int tid  = threadIdx.x;
  const int lane = tid & 31;
  const int wave = tid >> 5;
  const int wm = (wave >> 2) * 32;   // 0,32
  const int wn = (wave & 3) * 32;    // 0..96

  // per-thread copy slices: A 64x64 halves -> 32B/thread; B 128x64 -> 64B/thread
  const int aRow = tid >> 2, aCol = (tid & 3) << 4;
  const int bRow = tid >> 1, bCol = (tid & 1) << 5;
  const h16* gA = A  + (long)(blockIdx.x * BM + aRow) * lda + aCol;
  const h16* gB = Bt + (long)(blockIdx.y * BN + bRow) * ldb + bCol;
  const unsigned ldsA0 = lds_addr(&As[0][aRow][aCol]);
  const unsigned ldsA1 = lds_addr(&As[1][aRow][aCol]);
  const unsigned ldsB0 = lds_addr(&Bs[0][bRow][bCol]);
  const unsigned ldsB1 = lds_addr(&Bs[1][bRow][bCol]);

  v8f acc[2][2] = {};

  const int ktiles = K / BK;
  // prologue: stage tile 0 into buffer 0 (6 async ops per thread per stage)
  async_copy_32B(ldsA0, gA);
  async_copy_64B(ldsB0, gB);

  for (int kt = 0; kt < ktiles; ++kt) {
    const int buf = kt & 1;
    if (kt + 1 < ktiles) {
      // stream tile kt+1 into the other buffer while we compute on this one
      async_copy_32B(buf ? ldsA0 : ldsA1, gA + (kt + 1) * BK);
      async_copy_64B(buf ? ldsB0 : ldsB1, gB + (kt + 1) * BK);
      // 6 outstanding = only tile kt+1 in flight => tile kt complete (in-order)
      asm volatile("s_wait_asynccnt 6" ::: "memory");
    } else {
      asm volatile("s_wait_asynccnt 0" ::: "memory");
    }
    __syncthreads();   // all waves' copies for tile kt are visible

#pragma unroll
    for (int kk = 0; kk < BK; kk += 32) {
      v16h af0 = frag_a(&As[buf][wm][0],      kk, lane);
      v16h af1 = frag_a(&As[buf][wm + 16][0], kk, lane);
      v16h bf0 = frag_b(&Bs[buf][wn][0],      kk, lane);
      v16h bf1 = frag_b(&Bs[buf][wn + 16][0], kk, lane);
      acc[0][0] = __builtin_amdgcn_wmma_f32_16x16x32_f16(false, af0, false, bf0, (short)0, acc[0][0], false, false);
      acc[0][1] = __builtin_amdgcn_wmma_f32_16x16x32_f16(false, af0, false, bf1, (short)0, acc[0][1], false, false);
      acc[1][0] = __builtin_amdgcn_wmma_f32_16x16x32_f16(false, af1, false, bf0, (short)0, acc[1][0], false, false);
      acc[1][1] = __builtin_amdgcn_wmma_f32_16x16x32_f16(false, af1, false, bf1, (short)0, acc[1][1], false, false);
    }
    __syncthreads();   // done reading buf before it is overwritten at kt+2
  }

  // Epilogue (C/D layout: VGPR i -> row i + 8*(lane>=16), lanes%16 -> col)
  const int g  = lane >> 4, ln = lane & 15;
  const int mB = blockIdx.x * BM + wm;
  const int nB = blockIdx.y * BN + wn;
#pragma unroll
  for (int mt = 0; mt < 2; ++mt)
#pragma unroll
    for (int nt = 0; nt < 2; ++nt)
#pragma unroll
      for (int i = 0; i < 8; ++i) {
        const int row = mB + mt * 16 + g * 8 + i;
        const int col = nB + nt * 16 + ln;
        float v = acc[mt][nt][i];
        if (bias)     v += bias[col];
        if (rowScale) v *= rowScale[row];
        if (mode == 0)      ((float*)C)[cOff + (long)row * ldc + col] = v;
        else if (mode == 1) ((h16*)C)[cOff + (long)row * ldc + col] = (h16)v;
        else                ((h16*)C)[cOff + (long)col * ldc + row] = (h16)v;
      }
}

// ---------------------------------------------------------------------------
// Wave32 reductions
// ---------------------------------------------------------------------------
__device__ __forceinline__ float wave_sum(float v) {
#pragma unroll
  for (int m = 16; m; m >>= 1) v += __shfl_xor(v, m, 32);
  return v;
}
__device__ __forceinline__ float wave_max(float v) {
#pragma unroll
  for (int m = 16; m; m >>= 1) v = fmaxf(v, __shfl_xor(v, m, 32));
  return v;
}

// ---------------------------------------------------------------------------
// Elementwise / normalization kernels
// ---------------------------------------------------------------------------
__global__ __launch_bounds__(256)
void cast16_kernel(const float* __restrict__ s, h16* __restrict__ d, long n) {
  for (long i = (long)blockIdx.x * 256 + threadIdx.x; i < n; i += (long)gridDim.x * 256)
    d[i] = (h16)s[i];
}

// dst[b][c][r] = (f16) src[b][r][c]
__global__ __launch_bounds__(256)
void cast_tr_kernel(const float* __restrict__ src, h16* __restrict__ dst,
                    int rows, int cols) {
  const long n = (long)rows * cols;
  const float* s = src + (long)blockIdx.y * n;
  h16* d = dst + (long)blockIdx.y * n;
  for (long i = (long)blockIdx.x * 256 + threadIdx.x; i < n; i += (long)gridDim.x * 256) {
    const long r = i / cols, c = i - r * cols;
    d[c * rows + r] = (h16)s[i];
  }
}

// Softmax over E=512 of T16[h][b][s][e] * (mask[b,s] * E^-0.5); writes
// head16[b][s][h*512+e] (concat-over-heads layout for the led GEMM).
__global__ __launch_bounds__(256)
void softmax_kernel(const h16* __restrict__ T16, const float* __restrict__ mask,
                    h16* __restrict__ head) {
  const int r = blockIdx.x;             // h*8192 + b*1024 + s
  const int s = r & 1023;
  const int b = (r >> 10) & 7;
  const int h = r >> 13;
  const int tid = threadIdx.x;
  const float cs = mask[b * 1024 + s] * 0.04419417382415922f; // E^-0.5
  const h16* src = T16 + (long)r * 512;
  float v0 = (float)src[tid] * cs;
  float v1 = (float)src[tid + 256] * cs;

  __shared__ float red[8];
  float mx = wave_max(fmaxf(v0, v1));
  if ((tid & 31) == 0) red[tid >> 5] = mx;
  __syncthreads();
  mx = fmaxf(fmaxf(fmaxf(red[0], red[1]), fmaxf(red[2], red[3])),
             fmaxf(fmaxf(red[4], red[5]), fmaxf(red[6], red[7])));
  const float e0 = __expf(v0 - mx), e1 = __expf(v1 - mx);
  __syncthreads();
  float sm = wave_sum(e0 + e1);
  if ((tid & 31) == 0) red[tid >> 5] = sm;
  __syncthreads();
  const float tot = red[0] + red[1] + red[2] + red[3] + red[4] + red[5] + red[6] + red[7];
  const float inv = 1.0f / tot;
  h16* dst = head + ((long)(b * 1024 + s) * HE + h * 512);
  dst[tid]       = (h16)(e0 * inv);
  dst[tid + 256] = (h16)(e1 * inv);
}

// attnin[sg][h*512+e] = led2[sg][h*512+e] + q16[h][sg][e]   (in place on led2)
__global__ __launch_bounds__(256)
void add_qs_kernel(h16* __restrict__ led2, const h16* __restrict__ q16, long n) {
  for (long i = (long)blockIdx.x * 256 + threadIdx.x; i < n; i += (long)gridDim.x * 256) {
    const int  e  = (int)(i & 511);
    const int  h  = (int)((i >> 9) & 7);
    const long sg = i >> 12;
    led2[i] = (h16)((float)led2[i] + (float)q16[((long)h * TOK + sg) * 512 + e]);
  }
}

// exact GELU: 0.5*x*(1+erf(x/sqrt(2)))
__global__ __launch_bounds__(256)
void gelu_kernel(h16* __restrict__ p, long n) {
  for (long i = (long)blockIdx.x * 256 + threadIdx.x; i < n; i += (long)gridDim.x * 256) {
    const float x = (float)p[i];
    p[i] = (h16)(0.5f * x * (1.0f + erff(x * 0.7071067811865476f)));
  }
}

// out = LN(a+b)*g + be ; one block per row of 512; optional f16 copy
__global__ __launch_bounds__(256)
void add_ln_kernel(const float* __restrict__ a, const float* __restrict__ b,
                   const float* __restrict__ g, const float* __restrict__ be,
                   float* __restrict__ out, h16* __restrict__ out16) {
  const int row = blockIdx.x, tid = threadIdx.x;
  const long base = (long)row * 512;
  const float v0 = a[base + tid] + b[base + tid];
  const float v1 = a[base + tid + 256] + b[base + tid + 256];

  __shared__ float red[8];
  float s = wave_sum(v0 + v1);
  if ((tid & 31) == 0) red[tid >> 5] = s;
  __syncthreads();
  const float mu = (red[0] + red[1] + red[2] + red[3] + red[4] + red[5] + red[6] + red[7]) * (1.0f / 512.0f);
  const float d0 = v0 - mu, d1 = v1 - mu;
  __syncthreads();
  float s2 = wave_sum(d0 * d0 + d1 * d1);
  if ((tid & 31) == 0) red[tid >> 5] = s2;
  __syncthreads();
  const float var = (red[0] + red[1] + red[2] + red[3] + red[4] + red[5] + red[6] + red[7]) * (1.0f / 512.0f);
  const float inv = rsqrtf(var + 1e-5f);
  const float o0 = d0 * inv * g[tid] + be[tid];
  const float o1 = d1 * inv * g[tid + 256] + be[tid + 256];
  if (out)   { out[base + tid] = o0; out[base + tid + 256] = o1; }
  if (out16) { out16[base + tid] = (h16)o0; out16[base + tid + 256] = (h16)o1; }
}

// ---------------------------------------------------------------------------
// Host-side launch helpers
// ---------------------------------------------------------------------------
static inline void launch_gemm(hipStream_t st, const h16* A, const h16* Bt,
                               const float* bias, const float* rs, void* C,
                               int M, int N, int K, int lda, int ldb, int ldc,
                               long sA1, long sA2, long sB1, long sB2,
                               long sC1, long sC2, long sBias,
                               int z2count, int nz, int mode) {
  dim3 grid(M / BM, N / BN, nz);
  wmma_gemm_kernel<<<grid, 256, 0, st>>>(A, Bt, bias, rs, C, M, N, K, lda, ldb, ldc,
                                         sA1, sA2, sB1, sB2, sC1, sC2, sBias, z2count, mode);
}

static inline void launch_cast_tr(hipStream_t st, const float* src, h16* dst,
                                  int rows, int cols, int batch) {
  long n = (long)rows * cols;
  int gx = (int)((n + 255) / 256); if (gx > 8192) gx = 8192;
  cast_tr_kernel<<<dim3(gx, batch), 256, 0, st>>>(src, dst, rows, cols);
}

// ---------------------------------------------------------------------------
// Orchestration
// ---------------------------------------------------------------------------
extern "C" void kernel_launch(void* const* d_in, const int* in_sizes, int n_in,
                              void* d_out, int out_size, void* d_ws, size_t ws_size,
                              hipStream_t stream) {
  (void)in_sizes; (void)n_in; (void)out_size; (void)ws_size;
  const float* x    = (const float*)d_in[0];
  const float* mask = (const float*)d_in[1];
  const float* Wq1 = (const float*)d_in[2];  const float* bq1 = (const float*)d_in[3];
  const float* Wq2 = (const float*)d_in[4];  const float* bq2 = (const float*)d_in[5];
  const float* Wk1 = (const float*)d_in[6];  const float* bk1 = (const float*)d_in[7];
  const float* Wk2 = (const float*)d_in[8];  const float* bk2 = (const float*)d_in[9];
  const float* Wv1 = (const float*)d_in[10]; const float* bv1 = (const float*)d_in[11];
  const float* Wv2 = (const float*)d_in[12]; const float* bv2 = (const float*)d_in[13];
  const float* Wl1 = (const float*)d_in[14]; const float* bl1 = (const float*)d_in[15];
  const float* Wl2 = (const float*)d_in[16]; const float* bl2 = (const float*)d_in[17];
  const float* Wo  = (const float*)d_in[18]; const float* bo  = (const float*)d_in[19];
  const float* Ws1 = (const float*)d_in[20]; const float* bs1 = (const float*)d_in[21];
  const float* Ws2 = (const float*)d_in[22]; const float* bs2 = (const float*)d_in[23];
  const float* Wu1 = (const float*)d_in[24]; const float* bu1 = (const float*)d_in[25];
  const float* Wu2 = (const float*)d_in[26]; const float* bu2 = (const float*)d_in[27];
  const float* g1  = (const float*)d_in[28]; const float* be1 = (const float*)d_in[29];
  const float* g2  = (const float*)d_in[30]; const float* be2 = (const float*)d_in[31];

  // ---- workspace layout (bump allocator, 256B aligned) ----
  char* w = (char*)d_ws;
  size_t off = 0;
  auto alloc = [&](size_t bytes) -> void* {
    void* p = w + off; off += (bytes + 255) & ~(size_t)255; return p;
  };
  h16* X16   = (h16*)alloc((size_t)TOK * EDIM * 2);            // x in f16
  h16* WQ1T  = (h16*)alloc((size_t)HHEADS * EDIM * RDIM * 2);  // [h][r][e]
  h16* WQ2T  = (h16*)alloc((size_t)HHEADS * EDIM * RDIM * 2);  // [h][e][r]
  h16* WK1T  = (h16*)alloc((size_t)HHEADS * EDIM * RDIM * 2);
  h16* WK2T  = (h16*)alloc((size_t)HHEADS * EDIM * RDIM * 2);
  h16* WV1T  = (h16*)alloc((size_t)HHEADS * EDIM * RDIM * 2);
  h16* WV2T  = (h16*)alloc((size_t)HHEADS * EDIM * RDIM * 2);
  h16* WL1T  = (h16*)alloc((size_t)HE * RDIM * 2);             // [128][4096]
  h16* WL2T  = (h16*)alloc((size_t)HE * RDIM * 2);             // [4096][128]
  h16* WOT   = (h16*)alloc((size_t)HE * EDIM * 2);             // [512][4096]
  h16* WS1T  = (h16*)alloc((size_t)EDIM * RDIM * 2);           // [128][512]
  h16* WS2T  = (h16*)alloc((size_t)RDIM * FFDIM * 2);          // [2048][128]
  h16* WU1T  = (h16*)alloc((size_t)RDIM * FFDIM * 2);          // [128][2048]
  h16* WU2T  = (h16*)alloc((size_t)EDIM * RDIM * 2);           // [512][128]
  h16* HQ    = (h16*)alloc((size_t)HHEADS * TOK * RDIM * 2);   // stage-1 proj (reused q/k/v)
  h16* Q16   = (h16*)alloc((size_t)HHEADS * TOK * EDIM * 2);   // [h][sg][e]
  h16* K16T  = (h16*)alloc((size_t)HHEADS * TOK * EDIM * 2);   // [h][e][sg]; reused as T16
  h16* V16T  = (h16*)alloc((size_t)HHEADS * TOK * EDIM * 2);   // [h][e][sg]*m; reused as HEAD16
  h16* M16T  = (h16*)alloc((size_t)HHEADS * BB * EDIM * EDIM * 2); // [h][b][e2][e1]
  h16* LED1  = (h16*)alloc((size_t)TOK * RDIM * 2);
  h16* LED2  = (h16*)alloc((size_t)TOK * HE * 2);              // also attnin (in place)
  float* ATTN = (float*)alloc((size_t)TOK * EDIM * 4);         // reused as FF
  float* X1   = (float*)alloc((size_t)TOK * EDIM * 4);
  h16* X116  = (h16*)alloc((size_t)TOK * EDIM * 2);
  h16* HS16  = (h16*)alloc((size_t)TOK * RDIM * 2);
  h16* HF16  = (h16*)alloc((size_t)TOK * FFDIM * 2);
  h16* HU16  = (h16*)alloc((size_t)TOK * RDIM * 2);
  h16* T16   = K16T;   // [h][b][s][e]
  h16* HEAD  = V16T;   // [b][s][h*e]
  float* FF  = ATTN;

  // ---- stage inputs/weights into f16 (transposed weights) ----
  cast16_kernel<<<8192, 256, 0, stream>>>(x, X16, (long)TOK * EDIM);
  launch_cast_tr(stream, Wq1, WQ1T, EDIM, RDIM, HHEADS);
  launch_cast_tr(stream, Wq2, WQ2T, RDIM, EDIM, HHEADS);
  launch_cast_tr(stream, Wk1, WK1T, EDIM, RDIM, HHEADS);
  launch_cast_tr(stream, Wk2, WK2T, RDIM, EDIM, HHEADS);
  launch_cast_tr(stream, Wv1, WV1T, EDIM, RDIM, HHEADS);
  launch_cast_tr(stream, Wv2, WV2T, RDIM, EDIM, HHEADS);
  launch_cast_tr(stream, Wl1, WL1T, HE, RDIM, 1);
  launch_cast_tr(stream, Wl2, WL2T, RDIM, HE, 1);
  launch_cast_tr(stream, Wo,  WOT,  HE, EDIM, 1);
  launch_cast_tr(stream, Ws1, WS1T, EDIM, RDIM, 1);
  launch_cast_tr(stream, Ws2, WS2T, RDIM, FFDIM, 1);
  launch_cast_tr(stream, Wu1, WU1T, FFDIM, RDIM, 1);
  launch_cast_tr(stream, Wu2, WU2T, RDIM, EDIM, 1);

  const long sW = (long)EDIM * RDIM;          // per-head weight stride
  const long sHQ = (long)TOK * RDIM;          // per-head stage-1 stride
  const long sQ  = (long)TOK * EDIM;          // per-head q/kT/vT stride

  // ---- q = (x@Wq1+bq1)@Wq2+bq2  (batched over heads) ----
  launch_gemm(stream, X16, WQ1T, bq1, nullptr, HQ, TOK, RDIM, EDIM, EDIM, EDIM, RDIM,
              0, 0, sW, 0, sHQ, 0, RDIM, 1, HHEADS, 1);
  launch_gemm(stream, HQ, WQ2T, bq2, nullptr, Q16, TOK, EDIM, RDIM, RDIM, RDIM, EDIM,
              sHQ, 0, sW, 0, sQ, 0, EDIM, 1, HHEADS, 1);
  // ---- k (stored transposed [h][e][sg]) ----
  launch_gemm(stream, X16, WK1T, bk1, nullptr, HQ, TOK, RDIM, EDIM, EDIM, EDIM, RDIM,
              0, 0, sW, 0, sHQ, 0, RDIM, 1, HHEADS, 1);
  launch_gemm(stream, HQ, WK2T, bk2, nullptr, K16T, TOK, EDIM, RDIM, RDIM, RDIM, TOK,
              sHQ, 0, sW, 0, sQ, 0, EDIM, 1, HHEADS, 2);
  // ---- v (transposed, mask m_t folded in via rowScale) ----
  launch_gemm(stream, X16, WV1T, bv1, nullptr, HQ, TOK, RDIM, EDIM, EDIM, EDIM, RDIM,
              0, 0, sW, 0, sHQ, 0, RDIM, 1, HHEADS, 1);
  launch_gemm(stream, HQ, WV2T, bv2, mask, V16T, TOK, EDIM, RDIM, RDIM, RDIM, TOK,
              sHQ, 0, sW, 0, sQ, 0, EDIM, 1, HHEADS, 2);

  // ---- attention via associativity: M = K'^T V' (per b,h), then T = Q M ----
  launch_gemm(stream, K16T, V16T, nullptr, nullptr, M16T,
              EDIM, EDIM, SS, TOK, TOK, EDIM,
              sQ, SS, sQ, SS, (long)BB * EDIM * EDIM, (long)EDIM * EDIM,
              0, BB, HHEADS * BB, 2);
  launch_gemm(stream, Q16, M16T, nullptr, nullptr, T16,
              SS, EDIM, EDIM, EDIM, EDIM, EDIM,
              sQ, (long)SS * EDIM, (long)BB * EDIM * EDIM, (long)EDIM * EDIM,
              (long)BB * SS * EDIM, (long)SS * EDIM, 0, BB, HHEADS * BB, 1);

  // ---- softmax over feature axis (folds mask[s]*E^-0.5), concat heads ----
  softmax_kernel<<<HHEADS * TOK, 256, 0, stream>>>(T16, mask, HEAD);

  // ---- led path ----
  launch_gemm(stream, HEAD, WL1T, bl1, nullptr, LED1, TOK, RDIM, HE, HE, HE, RDIM,
              0, 0, 0, 0, 0, 0, 0, 1, 1, 1);
  launch_gemm(stream, LED1, WL2T, bl2, nullptr, LED2, TOK, HE, RDIM, RDIM, RDIM, HE,
              0, 0, 0, 0, 0, 0, 0, 1, 1, 1);
  add_qs_kernel<<<32768, 256, 0, stream>>>(LED2, Q16, (long)TOK * HE);
  launch_gemm(stream, LED2, WOT, bo, nullptr, ATTN, TOK, EDIM, HE, HE, HE, EDIM,
              0, 0, 0, 0, 0, 0, 0, 1, 1, 0);

  // ---- x1 = LN(attn + x) ----
  add_ln_kernel<<<TOK, 256, 0, stream>>>(ATTN, x, g1, be1, X1, X116);

  // ---- FFN squeeze/unsqueeze ----
  launch_gemm(stream, X116, WS1T, bs1, nullptr, HS16, TOK, RDIM, EDIM, EDIM, EDIM, RDIM,
              0, 0, 0, 0, 0, 0, 0, 1, 1, 1);
  launch_gemm(stream, HS16, WS2T, bs2, nullptr, HF16, TOK, FFDIM, RDIM, RDIM, RDIM, FFDIM,
              0, 0, 0, 0, 0, 0, 0, 1, 1, 1);
  gelu_kernel<<<32768, 256, 0, stream>>>(HF16, (long)TOK * FFDIM);
  launch_gemm(stream, HF16, WU1T, bu1, nullptr, HU16, TOK, RDIM, FFDIM, FFDIM, FFDIM, RDIM,
              0, 0, 0, 0, 0, 0, 0, 1, 1, 1);
  launch_gemm(stream, HU16, WU2T, bu2, nullptr, FF, TOK, EDIM, RDIM, RDIM, RDIM, EDIM,
              0, 0, 0, 0, 0, 0, 0, 1, 1, 0);

  // ---- out = LN(ff + x1) ----
  add_ln_kernel<<<TOK, 256, 0, stream>>>(FF, X1, g2, be2, (float*)d_out, nullptr);
}
